// UDFAGNNlayer_65386582114887
// MI455X (gfx1250) — matrix operand order repaired and checked by reading, not compiled, not run on hardware
//
#include <hip/hip_runtime.h>
#include <hip/hip_bf16.h>
#include <math.h>

typedef __attribute__((ext_vector_type(16))) _Float16 v16h;
typedef __attribute__((ext_vector_type(8)))  _Float16 v8h;
typedef __attribute__((ext_vector_type(2)))  _Float16 v2h;
typedef __attribute__((ext_vector_type(8)))  float    v8f;

#define FEAT 64  // feature dim (reference FEAT_DIM)

// ---------------------------------------------------------------------------
// Kernel 1: per-node inverse L2 norm + fp16 copy of raw features.
// One wave (32 lanes) per node; each lane owns 2 feature elements (float2).
// ---------------------------------------------------------------------------
__global__ void agnn_norm_kernel(const float* __restrict__ x,
                                 _Float16* __restrict__ xh,
                                 float* __restrict__ rnorm, int N) {
  const int lane = threadIdx.x & 31;
  const int wave = (blockIdx.x * blockDim.x + threadIdx.x) >> 5;
  if (wave >= N) return;
  const float2 v = ((const float2*)x)[(size_t)wave * (FEAT / 2) + lane];
  float ss = v.x * v.x + v.y * v.y;
#pragma unroll
  for (int k = 16; k >= 1; k >>= 1) ss += __shfl_xor(ss, k);
  if (lane == 0) rnorm[wave] = rsqrtf(ss + 1e-8f);
  v2h h; h.x = (_Float16)v.x; h.y = (_Float16)v.y;
  *(v2h*)(xh + (size_t)wave * FEAT + lane * 2) = h;
}

// ---------------------------------------------------------------------------
// Kernel 2: CSR row pointers from sorted row_index (binary search per node).
// ---------------------------------------------------------------------------
__global__ void agnn_rowptr_kernel(const int* __restrict__ rows,
                                   int* __restrict__ rowptr, int N, int E) {
  const int i = blockIdx.x * blockDim.x + threadIdx.x;
  if (i > N) return;
  int lo = 0, hi = E;
  while (lo < hi) {
    int mid = (lo + hi) >> 1;
    if (rows[mid] < i) lo = mid + 1; else hi = mid;
  }
  rowptr[i] = lo;
}

// ---------------------------------------------------------------------------
// Kernel 3: SDDMM via WMMA. One wave per 16 edges.
//   A(16x64 f16) = row-endpoint features, B(64x16 f16) = col-endpoint features
//   D = A x B  (two v_wmma_f32_16x16x32_f16, K=0..31 and K=32..63)
//   diag(D)[e] = dot(x[row_e], x[col_e]);  att = beta * rn_r * rn_c * diag
// VGPR layouts per CDNA5 ISA 7.12.2:
//   A 16-bit 16x32: lane m=lane&15; low half-wave K in {0..7,16..23},
//                   high half-wave K in {8..15,24..31} (two 16B chunks)
//   B 16-bit 32x16: lane n=lane&15; low half-wave K in {0..7,16..23},
//                   high half-wave K in {8..15,24..31} (two 16B chunks)
//   D 32-bit 16x16: lane group 0 -> M=v, group 1 -> M=v+8, N=lane&15
// ---------------------------------------------------------------------------
__global__ void agnn_sddmm_wmma_kernel(const _Float16* __restrict__ xh,
                                       const float* __restrict__ rnorm,
                                       const int* __restrict__ rows,
                                       const int* __restrict__ cols,
                                       const float* __restrict__ beta,
                                       float* __restrict__ att, int E) {
  const int lane  = threadIdx.x & 31;
  const long wave = (long)blockIdx.x * (blockDim.x >> 5) + (threadIdx.x >> 5);
  const long ebase = wave * 16;
  if (ebase >= E) return;  // wave-uniform exit

  const int  m  = lane & 15;
  const bool hi = lane >= 16;
  long e = ebase + m;
  if (e >= E) e = E - 1;  // clamp (duplicate work, write-guarded)
  const int r = rows[e];
  const int c = cols[e];

  const _Float16* pa = xh + (size_t)r * FEAT + (hi ? 8 : 0);
  const _Float16* pb = xh + (size_t)c * FEAT + (hi ? 8 : 0);

  v8f acc = {};
#pragma unroll
  for (int ko = 0; ko < FEAT; ko += 32) {
    v8h a0 = *(const v8h*)(pa + ko);        // A: K = basekA + 0..7
    v8h a1 = *(const v8h*)(pa + ko + 16);   // A: K = basekA + 16..23
    v8h b0 = *(const v8h*)(pb + ko);        // B: K = kb + 0..7
    v8h b1 = *(const v8h*)(pb + ko + 16);   // B: K = kb + 16..23
    v16h A, B;
#pragma unroll
    for (int j = 0; j < 8; ++j) {
      A[j] = a0[j]; A[j + 8] = a1[j];
      B[j] = b0[j]; B[j + 8] = b1[j];
    }
    acc = __builtin_amdgcn_wmma_f32_16x16x32_f16(
        /*neg_a=*/false, A, /*neg_b=*/false, B,
        /*c_mod=*/(short)0, acc, /*reuse_a=*/false, /*reuse_b=*/false);
  }

  // Extract diagonal D[m][m]: on this lane, VGPR v holds M = v + basek.
  const int basek = hi ? 8 : 0;
  float diag = 0.f;
#pragma unroll
  for (int v = 0; v < 8; ++v)
    if (v + basek == m) diag = acc[v];

  // Lanes 0..7 own edges 0..7 of the batch; lanes 24..31 own edges 8..15.
  if (m >= basek && m < basek + 8) {
    const long eo = ebase + m;
    if (eo < E) att[eo] = beta[0] * rnorm[r] * rnorm[c] * diag;
  }
}

// ---------------------------------------------------------------------------
// Kernel 4: per-row softmax + sparse aggregation. One wave per node.
// Lane owns 2 output feature elements; edge loop is wave-uniform.
// ---------------------------------------------------------------------------
__global__ void agnn_softmax_agg_kernel(const float* __restrict__ x,
                                        const int* __restrict__ cols,
                                        const float* __restrict__ att,
                                        const int* __restrict__ rowptr,
                                        float* __restrict__ out, int N) {
  const int lane = threadIdx.x & 31;
  const int node = (blockIdx.x * blockDim.x + threadIdx.x) >> 5;
  if (node >= N) return;
  const int s = rowptr[node];
  const int t = rowptr[node + 1];

  float mx = -INFINITY;
  for (int e = s + lane; e < t; e += 32) mx = fmaxf(mx, att[e]);
#pragma unroll
  for (int k = 16; k >= 1; k >>= 1) mx = fmaxf(mx, __shfl_xor(mx, k));

  float sum = 0.f;
  for (int e = s + lane; e < t; e += 32) sum += expf(att[e] - mx);
#pragma unroll
  for (int k = 16; k >= 1; k >>= 1) sum += __shfl_xor(sum, k);
  const float rden = 1.f / fmaxf(sum, 1e-16f);

  float2 acc = make_float2(0.f, 0.f);
  const float2* x2 = (const float2*)x;
  for (int e = s; e < t; ++e) {
    const float w = expf(att[e] - mx) * rden;  // broadcast scalar per edge
    const int c = cols[e];
    const float2 v = x2[(size_t)c * (FEAT / 2) + lane];
    acc.x += w * v.x;
    acc.y += w * v.y;
  }
  ((float2*)out)[(size_t)node * (FEAT / 2) + lane] = acc;  // also zeros empty rows
}

// ---------------------------------------------------------------------------
static inline size_t align256(size_t v) { return (v + 255) & ~(size_t)255; }

extern "C" void kernel_launch(void* const* d_in, const int* in_sizes, int n_in,
                              void* d_out, int out_size, void* d_ws, size_t ws_size,
                              hipStream_t stream) {
  (void)n_in; (void)out_size; (void)ws_size;
  const float* x    = (const float*)d_in[0];
  const float* beta = (const float*)d_in[1];
  const int* rows   = (const int*)d_in[2];
  const int* cols   = (const int*)d_in[3];
  float* out        = (float*)d_out;

  const int N = in_sizes[0] / FEAT;   // 50000
  const int E = in_sizes[2];          // 800000

  // Workspace layout
  char* ws = (char*)d_ws;
  size_t off = 0;
  _Float16* xh  = (_Float16*)(ws + off); off += align256((size_t)N * FEAT * sizeof(_Float16));
  float* rnorm  = (float*)(ws + off);    off += align256((size_t)N * sizeof(float));
  float* att    = (float*)(ws + off);    off += align256((size_t)E * sizeof(float));
  int* rowptr   = (int*)(ws + off);      off += align256((size_t)(N + 1) * sizeof(int));

  const int BLK = 256;                       // 8 wave32 per block
  const int wavesPerBlk = BLK / 32;

  // 1) norms + fp16 features (one wave per node)
  {
    int blocks = (N + wavesPerBlk - 1) / wavesPerBlk;
    agnn_norm_kernel<<<blocks, BLK, 0, stream>>>(x, xh, rnorm, N);
  }
  // 2) CSR row pointers (independent of 1)
  {
    int blocks = (N + 1 + BLK - 1) / BLK;
    agnn_rowptr_kernel<<<blocks, BLK, 0, stream>>>(rows, rowptr, N, E);
  }
  // 3) SDDMM edge attention via WMMA (one wave per 16 edges)
  {
    long waves = ((long)E + 15) / 16;
    long blocks = (waves + wavesPerBlk - 1) / wavesPerBlk;
    agnn_sddmm_wmma_kernel<<<(int)blocks, BLK, 0, stream>>>(xh, rnorm, rows, cols,
                                                            beta, att, E);
  }
  // 4) segment softmax + aggregation (one wave per node)
  {
    int blocks = (N + wavesPerBlk - 1) / wavesPerBlk;
    agnn_softmax_agg_kernel<<<blocks, BLK, 0, stream>>>(x, cols, att, rowptr, out, N);
  }
}
